// StaticDenoiseStep_30416958390314
// MI455X (gfx1250) — matrix operand order repaired and checked by reading, not compiled, not run on hardware
//
#include <hip/hip_runtime.h>
#include <math.h>

#define D_  1024
#define S_  50
#define B_  2
#define H_  8
#define DH_ 256
#define P_  1024
#define F_  4096
#define L_  6
#define BS_ (B_*S_)    /* 100 rows */
#define HD_ (H_*DH_)   /* 2048 */

typedef __attribute__((ext_vector_type(16))) __bf16 bf16x16;
typedef __attribute__((ext_vector_type(8)))  float  f32x8;

__device__ __forceinline__ float warp_sum32(float v) {
#pragma unroll
  for (int o = 16; o > 0; o >>= 1) v += __shfl_xor(v, o, 32);
  return v;
}

// ---------------- time embedding: temb (B, D) ----------------
__global__ void time_embed_kernel(const float* __restrict__ t, float* __restrict__ temb) {
  int idx = blockIdx.x * blockDim.x + threadIdx.x;
  if (idx >= B_ * D_) return;
  int b = idx / D_, j = idx % D_;
  int i = (j < D_ / 2) ? j : j - D_ / 2;
  float frac = (float)i / (float)(D_ / 2 - 1);
  float period = 0.004f * __powf(1000.0f, frac);     // 0.004*(4/0.004)^frac
  float ang = (6.283185307179586f / period) * t[b];
  temb[idx] = (j < D_ / 2) ? __sinf(ang) : __cosf(ang);
}

// ---------------- 2-row GEMV:  Y(2,N) = act( X(2,K) @ W(K,N) + bias ) ----------------
// block = 256 = 64 cols x 4 K-groups; W rows are coalesced across the 64 cols.
__global__ void gemv2_kernel(const float* __restrict__ X, const float* __restrict__ W,
                             const float* __restrict__ bias, float* __restrict__ Y,
                             int K, int N, int act) {
  __shared__ float red[4][2][64];
  int cb = threadIdx.x & 63;
  int kg = threadIdx.x >> 6;
  int c  = blockIdx.x * 64 + cb;
  int kpg = K >> 2;
  float a0 = 0.f, a1 = 0.f;
  const float* wp = W + (size_t)kg * kpg * N + c;
  const float* x0 = X + kg * kpg;
  const float* x1 = X + K + kg * kpg;
  for (int k = 0; k < kpg; ++k) {
    float w = wp[(size_t)k * N];
    a0 = fmaf(x0[k], w, a0);
    a1 = fmaf(x1[k], w, a1);
  }
  red[kg][0][cb] = a0;
  red[kg][1][cb] = a1;
  __syncthreads();
  if (kg == 0) {
    float s0 = red[0][0][cb] + red[1][0][cb] + red[2][0][cb] + red[3][0][cb];
    float s1 = red[0][1][cb] + red[1][1][cb] + red[2][1][cb] + red[3][1][cb];
    if (bias) { s0 += bias[c]; s1 += bias[c]; }
    if (act == 1) {                      // silu
      s0 = s0 / (1.f + __expf(-s0));
      s1 = s1 / (1.f + __expf(-s1));
    }
    Y[c] = s0;
    Y[N + c] = s1;
  }
}

// ---------------- adaRMS: out = rms(h)*(1+scale) + shift ; mod = (B, 3D) ----------------
__global__ void ada_rms_kernel(const float* __restrict__ h, const float* __restrict__ mod,
                               float* __restrict__ out) {
  __shared__ float sh[256];
  int row = blockIdx.x;                  // 0..99
  int b = row / S_;
  const float* hp = h + (size_t)row * D_;
  float ss = 0.f;
  for (int d = threadIdx.x; d < D_; d += 256) { float x = hp[d]; ss = fmaf(x, x, ss); }
  sh[threadIdx.x] = ss;
  __syncthreads();
  for (int o = 128; o > 0; o >>= 1) {
    if (threadIdx.x < o) sh[threadIdx.x] += sh[threadIdx.x + o];
    __syncthreads();
  }
  float rms = rsqrtf(sh[0] / (float)D_ + 1e-6f);
  const float* sc = mod + (size_t)b * 3 * D_;
  const float* sf = sc + D_;
  float* op = out + (size_t)row * D_;
  for (int d = threadIdx.x; d < D_; d += 256)
    op[d] = hp[d] * rms * (1.f + sc[d]) + sf[d];
}

// ---------------- bf16 WMMA GEMM: C(M,N) = A(M,K) @ W(K,N) [+bias] ----------------
// block = 128 (4 waves); each wave owns a 16-wide N slice and loops all MT m-tiles,
// so W is streamed from HBM exactly once. K stepped by 32 (one v_wmma per tile).
// MT is a compile-time constant so the 7x v8f accumulators stay in VGPRs (no
// dynamically-indexed array -> no scratch spill around the WMMAs).
template <int MT>
__global__ void gemm_wmma_kernel(const float* __restrict__ A, const float* __restrict__ W,
                                 const float* __restrict__ bias, float* __restrict__ C,
                                 int M, int K, int N) {
  int wave = threadIdx.x >> 5;
  int lane = threadIdx.x & 31;
  int half = lane >> 4;
  int lidx = lane & 15;
  int n0 = blockIdx.x * 64 + wave * 16;
  if (n0 >= N) return;                   // wave-uniform
  int ncol = n0 + lidx; if (ncol >= N) ncol = N - 1;

  f32x8 acc[MT];
#pragma unroll
  for (int i = 0; i < MT; ++i)
#pragma unroll
    for (int v = 0; v < 8; ++v) acc[i][v] = 0.f;

  // Clamped per-m-tile A row pointers, hoisted out of the K loop.
  const float* arow[MT];
#pragma unroll
  for (int mt = 0; mt < MT; ++mt) {
    int row = mt * 16 + lidx; if (row >= M) row = M - 1;
    arow[mt] = A + (size_t)row * K;
  }

  // A-frag per-lane K offsets and B-frag K base: contiguous k = 16*half + e for both operands
  int akoff[16];
#pragma unroll
  for (int e = 0; e < 16; ++e) akoff[e] = half * 16 + e;
  int bk0 = half * 16;

  const float* wcol = W + (size_t)bk0 * N + ncol;
  for (int kk = 0; kk < K; kk += 32) {
    bf16x16 bf;
#pragma unroll
    for (int e = 0; e < 16; ++e)
      bf[e] = (__bf16)wcol[(size_t)e * N];
    wcol += (size_t)32 * N;
#pragma unroll
    for (int mt = 0; mt < MT; ++mt) {
      const float* ap = arow[mt] + kk;
      bf16x16 af;
#pragma unroll
      for (int e = 0; e < 16; ++e) af[e] = (__bf16)ap[akoff[e]];
      acc[mt] = __builtin_amdgcn_wmma_f32_16x16x32_bf16(
          false, af, false, bf, (short)0, acc[mt], false, false);
    }
  }

  float bv = bias ? bias[ncol] : 0.f;
  int n = n0 + lidx;
#pragma unroll
  for (int mt = 0; mt < MT; ++mt) {
#pragma unroll
    for (int v = 0; v < 8; ++v) {
      int m = mt * 16 + half * 8 + v;
      if (m < M && n < N) C[(size_t)m * N + n] = acc[mt][v] + bv;
    }
  }
}

// ---------------- rope (gemma, base 10000) applied in place to (100, 2048) ----------------
__global__ void rope_kernel(float* __restrict__ buf) {
  int idx = blockIdx.x * blockDim.x + threadIdx.x;   // pair index
  if (idx >= BS_ * H_ * (DH_ / 2)) return;
  int row = idx / (H_ * (DH_ / 2));
  int r = idx % (H_ * (DH_ / 2));
  int hh = r / (DH_ / 2);
  int i = r % (DH_ / 2);
  float pos = (float)(P_ + (row % S_));
  float freq = pos * __powf(10000.0f, -(float)i / (float)(DH_ / 2));
  float c = __cosf(freq), sn = __sinf(freq);
  size_t base = (size_t)row * HD_ + hh * DH_;
  float x1 = buf[base + i], x2 = buf[base + DH_ / 2 + i];
  buf[base + i]            = x1 * c - x2 * sn;
  buf[base + DH_ / 2 + i]  = x2 * c + x1 * sn;
}

// ---------------- attention: one wave per (b,h,s); online softmax in fp32 ----------------
// prefix K/V read straight from the cache (layer-offset pointers), suffix from q/k/v bufs.
__global__ void attn_kernel(const float* __restrict__ qb, const float* __restrict__ kb,
                            const float* __restrict__ vb, const float* __restrict__ ck,
                            const float* __restrict__ cv, float* __restrict__ ob) {
  int wid = (blockIdx.x * blockDim.x + threadIdx.x) >> 5;
  if (wid >= B_ * H_ * S_) return;
  int lane = threadIdx.x & 31;
  int b = wid / (H_ * S_);
  int rem = wid % (H_ * S_);
  int hh = rem / S_;
  int s = rem % S_;

  const float4* q4 = reinterpret_cast<const float4*>(
      qb + ((size_t)(b * S_ + s)) * HD_ + hh * DH_ + lane * 8);
  float4 qa = q4[0], qb4 = q4[1];

  float m = -1e30f, lsum = 0.f;
  float acc[8];
#pragma unroll
  for (int i = 0; i < 8; ++i) acc[i] = 0.f;

  int Tmax = P_ + s + 1;                 // prefix fully visible, suffix causal
  for (int t = 0; t < Tmax; ++t) {
    const float4* k4;
    const float4* v4;
    if (t < P_) {
      size_t o = (((size_t)(b * H_ + hh)) * P_ + t) * DH_ + lane * 8;
      k4 = reinterpret_cast<const float4*>(ck + o);
      v4 = reinterpret_cast<const float4*>(cv + o);
    } else {
      size_t o = ((size_t)(b * S_ + (t - P_))) * HD_ + hh * DH_ + lane * 8;
      k4 = reinterpret_cast<const float4*>(kb + o);
      v4 = reinterpret_cast<const float4*>(vb + o);
    }
    float4 ka = k4[0], kb2 = k4[1];
    float d = qa.x * ka.x + qa.y * ka.y + qa.z * ka.z + qa.w * ka.w +
              qb4.x * kb2.x + qb4.y * kb2.y + qb4.z * kb2.z + qb4.w * kb2.w;
    d = warp_sum32(d);
    float score = d * 0.0625f;           // DH^-0.5
    float mn = fmaxf(m, score);
    float corr = __expf(m - mn);
    float p = __expf(score - mn);
    lsum = lsum * corr + p;
    float4 va = v4[0], vb2 = v4[1];
    acc[0] = acc[0] * corr + p * va.x;  acc[1] = acc[1] * corr + p * va.y;
    acc[2] = acc[2] * corr + p * va.z;  acc[3] = acc[3] * corr + p * va.w;
    acc[4] = acc[4] * corr + p * vb2.x; acc[5] = acc[5] * corr + p * vb2.y;
    acc[6] = acc[6] * corr + p * vb2.z; acc[7] = acc[7] * corr + p * vb2.w;
    m = mn;
  }
  float inv = 1.f / lsum;
  float* op = ob + ((size_t)(b * S_ + s)) * HD_ + hh * DH_ + lane * 8;
#pragma unroll
  for (int i = 0; i < 8; ++i) op[i] = acc[i] * inv;
}

// ---------------- gelu(tanh) * up ----------------
__global__ void gelu_mul_kernel(const float* __restrict__ g, const float* __restrict__ u,
                                float* __restrict__ out, int n) {
  int idx = blockIdx.x * blockDim.x + threadIdx.x;
  if (idx >= n) return;
  float x = g[idx];
  float t = tanhf(0.7978845608028654f * (x + 0.044715f * x * x * x));
  out[idx] = 0.5f * x * (1.f + t) * u[idx];
}

// ---------------- gated residual: h += gate_b * x ----------------
__global__ void gated_res_kernel(float* __restrict__ h, const float* __restrict__ mod,
                                 const float* __restrict__ x) {
  int idx = blockIdx.x * blockDim.x + threadIdx.x;
  if (idx >= BS_ * D_) return;
  int row = idx / D_, d = idx % D_;
  int b = row / S_;
  h[idx] = h[idx] + mod[(size_t)b * 3 * D_ + 2 * D_ + d] * x[idx];
}

extern "C" void kernel_launch(void* const* d_in, const int* in_sizes, int n_in,
                              void* d_out, int out_size, void* d_ws, size_t ws_size,
                              hipStream_t stream) {
  (void)in_sizes; (void)n_in; (void)out_size; (void)ws_size;
  const float* x_t       = (const float*)d_in[1];
  const float* timestep  = (const float*)d_in[2];
  const float* cached_k  = (const float*)d_in[3];
  const float* cached_v  = (const float*)d_in[4];
  const float* w_time_in = (const float*)d_in[5];
  const float* b_time_in = (const float*)d_in[6];
  const float* w_time_out= (const float*)d_in[7];
  const float* b_time_out= (const float*)d_in[8];
  const float* w_act_in  = (const float*)d_in[9];
  const float* b_act_in  = (const float*)d_in[10];
  const float* w_act_out = (const float*)d_in[11];
  const float* b_act_out = (const float*)d_in[12];
  const float* wq        = (const float*)d_in[13];
  const float* wk        = (const float*)d_in[14];
  const float* wv        = (const float*)d_in[15];
  const float* wo        = (const float*)d_in[16];
  const float* w_gate    = (const float*)d_in[17];
  const float* w_up      = (const float*)d_in[18];
  const float* w_down    = (const float*)d_in[19];
  const float* w_norm1   = (const float*)d_in[20];
  const float* w_norm2   = (const float*)d_in[21];
  const float* w_norm_f  = (const float*)d_in[22];

  // workspace layout (floats)
  float* w = (float*)d_ws;
  size_t off = 0;
  auto alloc = [&](size_t n) { float* p = w + off; off += n; return p; };
  float* temb = alloc(B_ * D_);
  float* th   = alloc(B_ * D_);
  float* cond = alloc(B_ * D_);
  float* mod  = alloc(B_ * 3 * D_);
  float* hbuf = alloc((size_t)BS_ * D_);
  float* nbuf = alloc((size_t)BS_ * D_);
  float* qbuf = alloc((size_t)BS_ * HD_);
  float* kbuf = alloc((size_t)BS_ * HD_);
  float* vbuf = alloc((size_t)BS_ * HD_);
  float* abuf = alloc((size_t)BS_ * HD_);
  float* pbuf = alloc((size_t)BS_ * D_);
  float* gbuf = alloc((size_t)BS_ * F_);
  float* ubuf = alloc((size_t)BS_ * F_);
  float* mbuf = alloc((size_t)BS_ * F_);

  auto gemv = [&](const float* X, const float* W_, const float* bias, float* Y,
                  int K, int N, int act) {
    gemv2_kernel<<<dim3(N / 64), 256, 0, stream>>>(X, W_, bias, Y, K, N, act);
  };
  // M = 100 for every GEMM -> MT = 7 m-tiles per wave (compile-time).
  auto gemm = [&](const float* A, const float* W_, const float* bias, float* C,
                  int M, int K, int N) {
    gemm_wmma_kernel<7><<<dim3((N + 63) / 64), 128, 0, stream>>>(A, W_, bias, C, M, K, N);
  };

  // ---- time embedding + time MLP -> cond ----
  time_embed_kernel<<<dim3((B_ * D_ + 255) / 256), 256, 0, stream>>>(timestep, temb);
  gemv(temb, w_time_in, b_time_in, th, D_, D_, 1);
  gemv(th, w_time_out, b_time_out, cond, D_, D_, 1);

  // ---- action embedding: h = x_t @ w_act_in + b ----
  gemm(x_t, w_act_in, b_act_in, hbuf, BS_, 32, D_);

  const size_t wqkv_stride = (size_t)D_ * HD_;          // 1024*2048
  const size_t wnorm_stride = (size_t)D_ * 3 * D_;      // 1024*3072
  const size_t wffn_stride = (size_t)D_ * F_;           // 1024*4096
  const size_t cache_stride = (size_t)B_ * H_ * P_ * DH_;

  for (int l = 0; l < L_; ++l) {
    // attn block
    gemv(cond, w_norm1 + l * wnorm_stride, nullptr, mod, D_, 3 * D_, 0);
    ada_rms_kernel<<<dim3(BS_), 256, 0, stream>>>(hbuf, mod, nbuf);
    gemm(nbuf, wq + l * wqkv_stride, nullptr, qbuf, BS_, D_, HD_);
    gemm(nbuf, wk + l * wqkv_stride, nullptr, kbuf, BS_, D_, HD_);
    gemm(nbuf, wv + l * wqkv_stride, nullptr, vbuf, BS_, D_, HD_);
    int rpairs = BS_ * H_ * (DH_ / 2);
    rope_kernel<<<dim3((rpairs + 255) / 256), 256, 0, stream>>>(qbuf);
    rope_kernel<<<dim3((rpairs + 255) / 256), 256, 0, stream>>>(kbuf);
    attn_kernel<<<dim3((B_ * H_ * S_) / 8), 256, 0, stream>>>(
        qbuf, kbuf, vbuf, cached_k + l * cache_stride, cached_v + l * cache_stride, abuf);
    gemm(abuf, wo + l * wqkv_stride, nullptr, pbuf, BS_, HD_, D_);
    gated_res_kernel<<<dim3((BS_ * D_ + 255) / 256), 256, 0, stream>>>(hbuf, mod, pbuf);

    // mlp block
    gemv(cond, w_norm2 + l * wnorm_stride, nullptr, mod, D_, 3 * D_, 0);
    ada_rms_kernel<<<dim3(BS_), 256, 0, stream>>>(hbuf, mod, nbuf);
    gemm(nbuf, w_gate + l * wffn_stride, nullptr, gbuf, BS_, D_, F_);
    gemm(nbuf, w_up + l * wffn_stride, nullptr, ubuf, BS_, D_, F_);
    gelu_mul_kernel<<<dim3((BS_ * F_ + 255) / 256), 256, 0, stream>>>(gbuf, ubuf, mbuf, BS_ * F_);
    gemm(mbuf, w_down + l * wffn_stride, nullptr, pbuf, BS_, F_, D_);
    gated_res_kernel<<<dim3((BS_ * D_ + 255) / 256), 256, 0, stream>>>(hbuf, mod, pbuf);
  }

  // ---- final adaRMS + out projection ----
  gemv(cond, w_norm_f, nullptr, mod, D_, 3 * D_, 0);
  ada_rms_kernel<<<dim3(BS_), 256, 0, stream>>>(hbuf, mod, nbuf);
  gemm(nbuf, w_act_out, b_act_out, (float*)d_out, BS_, D_, 32);
}
